// WaveDensityAttentionBlock_17051020165570
// MI455X (gfx1250) — compile-verified, matrix operand order checked by reading, not run on hardware
//
#include <hip/hip_runtime.h>
#include <math.h>

// ---------------- CDNA5 (gfx1250) wave32 WMMA types ----------------
typedef __attribute__((ext_vector_type(16))) __bf16 v16bf;
typedef __attribute__((ext_vector_type(8)))  float  v8f;

#define DMODEL 1024
#define SEQ    1024
#define NB     2
#define NH     16
#define HD     64
#define NM     16
#define NW     8
#define NRR    8
#define NCR    8
#define CKPAD  48

__device__ __forceinline__ v8f wmma_bf16(v16bf a, v16bf b, v8f c) {
    return __builtin_amdgcn_wmma_f32_16x16x32_bf16(false, a, false, b, (short)0, c, false, false);
}

// ---------------- small helper kernels ----------------
// vectorized fp32 -> bf16 (n multiple of 4)
__global__ __launch_bounds__(256) void cvt_bf16_k(const float* __restrict__ in, __bf16* __restrict__ out, int n4) {
    int i = blockIdx.x * 256 + threadIdx.x;
    if (i < n4) {
        float4 v = ((const float4*)in)[i];
        union { __bf16 e[4]; uint2 u; } o;
        o.e[0] = (__bf16)v.x; o.e[1] = (__bf16)v.y; o.e[2] = (__bf16)v.z; o.e[3] = (__bf16)v.w;
        ((uint2*)out)[i] = o.u;
    }
}

// vectorized bf16 zero (n multiple of 8)
__global__ __launch_bounds__(256) void zero_bf16_k(__bf16* __restrict__ out, int n8) {
    int i = blockIdx.x * 256 + threadIdx.x;
    if (i < n8) {
        uint4 z; z.x = z.y = z.z = z.w = 0u;
        ((uint4*)out)[i] = z;
    }
}

// pooled[b,d] = mean_s x[b,s,d]   (grid = B*D/256)
__global__ __launch_bounds__(256) void pool_mean_k(const float* __restrict__ x, float* __restrict__ pooled) {
    int idx = blockIdx.x * 256 + threadIdx.x;      // B*D = 2048
    int b = idx >> 10, d = idx & 1023;
    float s = 0.f;
    for (int ss = 0; ss < SEQ; ++ss) s += x[((size_t)b * SEQ + ss) * DMODEL + d];
    pooled[idx] = s * (1.f / SEQ);
}

// gating: gl = pooled@gate_w + b + mod@mod_basis ; wts = softmax(gl) per (b,h). grid = B.
__global__ __launch_bounds__(256) void gating_k(const float* __restrict__ pooled,
                                                const float* __restrict__ gate_w, const float* __restrict__ gate_b,
                                                const float* __restrict__ mod_w,  const float* __restrict__ mod_b,
                                                const float* __restrict__ mod_basis,
                                                float* __restrict__ wts) {
    __shared__ float modS[NH * NRR];
    __shared__ float glS[NH * NM];
    int b = blockIdx.x, t = threadIdx.x;
    const float* pb = pooled + (size_t)b * DMODEL;
    if (t < NH * NRR) {
        float a = mod_b[t];
        for (int d = 0; d < DMODEL; ++d) a += pb[d] * mod_w[(size_t)d * (NH * NRR) + t];
        modS[t] = a;
    }
    __syncthreads();
    if (t < NH * NM) {
        float a = gate_b[t];
        for (int d = 0; d < DMODEL; ++d) a += pb[d] * gate_w[(size_t)d * (NH * NM) + t];
        int h = t >> 4, m = t & 15;
        for (int r = 0; r < NRR; ++r) a += modS[h * NRR + r] * mod_basis[((size_t)r * NH + h) * NM + m];
        glS[t] = a;
    }
    __syncthreads();
    if (t < NH) {
        float mx = -1e30f;
        for (int m = 0; m < NM; ++m) mx = fmaxf(mx, glS[t * NM + m]);
        float e[NM]; float sm = 0.f;
        for (int m = 0; m < NM; ++m) { e[m] = __expf(glS[t * NM + m] - mx); sm += e[m]; }
        float inv = 1.f / sm;
        for (int m = 0; m < NM; ++m) wts[((size_t)b * NH + t) * NM + m] = e[m] * inv;
    }
}

// causal_kern[b,h,d] = ALPHA * sigmoid(6*kern) * window(d).  grid = B*H*S/256.
__global__ __launch_bounds__(256) void wave_density_k(const float* __restrict__ freqs, const float* __restrict__ amps,
                                                      const float* __restrict__ phases, const float* __restrict__ wts,
                                                      float* __restrict__ ckern) {
    int idx = blockIdx.x * 256 + threadIdx.x;      // B*H*S
    int d = idx & (SEQ - 1);
    int bh = idx >> 10;
    int h = bh & (NH - 1);
    float rel = (float)d;
    float p1 = rel * (1.f / SEQ);
    float p2 = (d == 0) ? 0.f : sqrtf(rel + 1e-6f) * (1.f / 32.f);   // sqrt(S)=32
    float kern = 0.f;
    for (int m = 0; m < NM; ++m) {
        const float* fq = freqs  + ((size_t)(h * NM + m) * NW) * 2;
        const float* am = amps   +  (size_t)(h * NM + m) * NW;
        const float* ph = phases +  (size_t)(h * NM + m) * NW;
        float acc = 0.f;
        #pragma unroll
        for (int w = 0; w < NW; ++w) {
            float arg = fq[2 * w] * p1 + fq[2 * w + 1] * p2 + ph[w];
            acc += am[w] * __sinf(arg);
        }
        kern += wts[(size_t)bh * NM + m] * acc;
    }
    float dens = 1.f / (1.f + __expf(-6.f * kern));          // straight-through forward = sigmoid
    float wind = __expf(-(rel * rel) * (1.f / 131072.f));    // 2*(S/4)^2
    ckern[idx] = 3.0f * dens * wind;
}

// qm/km[b,h,s] token modulation.  grid = B*S.
__global__ __launch_bounds__(256) void qkmod_k(const float* __restrict__ x,
                                               const float* __restrict__ qw, const float* __restrict__ qb,
                                               const float* __restrict__ kw, const float* __restrict__ kb,
                                               float* __restrict__ qm, float* __restrict__ km) {
    __shared__ float xrow[DMODEL];
    __shared__ float red[32][9];
    int bs = blockIdx.x; int b = bs >> 10, s = bs & 1023;
    int t = threadIdx.x;
    for (int i = t; i < DMODEL; i += 256) xrow[i] = x[(size_t)bs * DMODEL + i];
    __syncthreads();
    int o = t & 31, sl = t >> 5;
    int h = o & 15; bool isK = o >= 16;
    const float* wcol = isK ? kw : qw;
    float a = 0.f;
    for (int d = sl * 128; d < sl * 128 + 128; ++d) a += xrow[d] * wcol[(size_t)d * NH + h];
    red[o][sl] = a;
    __syncthreads();
    if (t < 32) {
        float acc = 0.f;
        #pragma unroll
        for (int i = 0; i < 8; ++i) acc += red[t][i];
        int hh = t & 15;
        if (t < 16) qm[((size_t)b * NH + hh) * SEQ + s] = acc + qb[hh];
        else        km[((size_t)b * NH + hh) * SEQ + s] = acc + kb[hh];
    }
}

// ---------------- generic bf16 WMMA GEMM with fused epilogues ----------------
// EPI 1: out = bf16(acc + bias[n]) -> V^T layout [B,H,HD,S]   (keys contiguous for attention)
// EPI 2: out = bf16(acc)           -> padded qc/kc layout [B,H,S,32] (cols 0..7 valid)
// EPI 3: outF = acc + bias[n] + resid[m,n]    (fp32)
// EPI 4: outB = bf16(gelu(acc + bias[n]))     ([M,N] bf16)
template <int EPI>
__global__ __launch_bounds__(256)
void gemm_bf16_k(const __bf16* __restrict__ A, const __bf16* __restrict__ B,
                 int M, int N, int K,
                 const float* __restrict__ bias, const float* __restrict__ resid,
                 float* __restrict__ outF, __bf16* __restrict__ outB) {
    constexpr int BM = 128, BN = 128, BK = 32, LK = 48;     // LK keeps b128 alignment, spreads banks
    __shared__ alignas(16) __bf16 As[BM * LK];
    __shared__ alignas(16) __bf16 Bs[BN * LK];
    const int tid = threadIdx.x;
    const int lane = tid & 31, wave = tid >> 5;
    const int half = lane >> 4, l16 = lane & 15;
    const int wm = wave >> 2, wn = wave & 3;                // 2 x 4 wave grid
    const int m0 = blockIdx.y * BM, n0 = blockIdx.x * BN;

    v8f acc[4][2];
    #pragma unroll
    for (int i = 0; i < 4; ++i)
        #pragma unroll
        for (int j = 0; j < 2; ++j)
            #pragma unroll
            for (int e = 0; e < 8; ++e) acc[i][j][e] = 0.f;

    for (int k0 = 0; k0 < K; k0 += BK) {
        #pragma unroll
        for (int it = 0; it < 2; ++it) {                    // A: 128x32, row-major stage
            int c = tid + it * 256;
            int row = c >> 2, col = (c & 3) * 8;
            uint4 va = *(const uint4*)(A + (size_t)(m0 + row) * K + k0 + col);
            *(uint4*)(As + row * LK + col) = va;
        }
        {                                                   // B: 32x128 -> transposed [n][k], 2 k-rows/thread
            int kk = (tid >> 4) * 2, nc = (tid & 15) * 8;
            uint4 b0 = *(const uint4*)(B + (size_t)(k0 + kk) * N + n0 + nc);
            uint4 b1 = *(const uint4*)(B + (size_t)(k0 + kk + 1) * N + n0 + nc);
            const __bf16* p0 = (const __bf16*)&b0;
            const __bf16* p1 = (const __bf16*)&b1;
            #pragma unroll
            for (int e = 0; e < 8; ++e) {
                union { __bf16 e2[2]; uint u; } t;
                t.e2[0] = p0[e]; t.e2[1] = p1[e];
                *(uint*)(Bs + (nc + e) * LK + kk) = t.u;    // kk even -> 4B aligned
            }
        }
        __syncthreads();

        v16bf afr[4];
        #pragma unroll
        for (int mt = 0; mt < 4; ++mt) {                    // A frag: lane=M, K halves per lane-half
            int row = wm * 64 + mt * 16 + l16;
            union { v16bf v; uint4 q[2]; } u;
            u.q[0] = *(const uint4*)(As + row * LK + half * 8);
            u.q[1] = *(const uint4*)(As + row * LK + 16 + half * 8);
            afr[mt] = u.v;
        }
        #pragma unroll
        for (int nt = 0; nt < 2; ++nt) {                    // B frag: lane=N, same K mapping
            int col = wn * 32 + nt * 16 + l16;
            union { v16bf v; uint4 q[2]; } u;
            u.q[0] = *(const uint4*)(Bs + col * LK + half * 8);
            u.q[1] = *(const uint4*)(Bs + col * LK + 16 + half * 8);
            v16bf bfr = u.v;
            #pragma unroll
            for (int mt = 0; mt < 4; ++mt)
                acc[mt][nt] = wmma_bf16(afr[mt], bfr, acc[mt][nt]);
        }
        __syncthreads();
    }

    #pragma unroll
    for (int mt = 0; mt < 4; ++mt)
        #pragma unroll
        for (int nt = 0; nt < 2; ++nt)
            #pragma unroll
            for (int r = 0; r < 8; ++r) {                   // C: lane=N, row = r + 8*half
                int gm = m0 + wm * 64 + mt * 16 + r + half * 8;
                int gn = n0 + wn * 32 + nt * 16 + l16;
                float v = acc[mt][nt][r];
                if (EPI == 1) {
                    float val = v + bias[gn];
                    int b = gm >> 10, s = gm & 1023, h = gn >> 6, hd = gn & 63;
                    outB[(((size_t)b * NH + h) * HD + hd) * SEQ + s] = (__bf16)val;   // V^T
                } else if (EPI == 2) {
                    int b = gm >> 10, s = gm & 1023, h = gn >> 3, cc = gn & 7;
                    outB[(((size_t)b * NH + h) * SEQ + s) * 32 + cc] = (__bf16)v;
                } else if (EPI == 3) {
                    size_t o = (size_t)gm * N + gn;
                    outF[o] = v + bias[gn] + resid[o];
                } else if (EPI == 4) {
                    float xg = v + bias[gn];
                    float g = 0.5f * xg * (1.0f + erff(xg * 0.70710678f));
                    outB[(size_t)gm * N + gn] = (__bf16)g;
                }
            }
}

// ---------------- streaming wave-density attention ----------------
// grid = (S/128, H, B); 8 waves, each owns a 16-row q tile; 32 keys/step; 6 WMMAs/step.
// V is consumed transposed: vT[b,h,hd,s] so every B-fragment is two aligned b128 loads.
// ck_pad has a CKPAD zero guard band so Toeplitz gathers are base + immediate-offset DS loads.
__global__ __launch_bounds__(256)
void attn_k(const __bf16* __restrict__ qcp, const __bf16* __restrict__ kcp,
            const __bf16* __restrict__ vT,
            const float* __restrict__ ckern, const float* __restrict__ qm,
            const float* __restrict__ km, __bf16* __restrict__ ctxb) {
    __shared__ float ck_pad[CKPAD + SEQ];
    __shared__ float km_s[SEQ];
    __shared__ float qm_s[128];
    __shared__ float sS[8][16][33];
    __shared__ alignas(16) __bf16 pS[8][16][32];
    __shared__ float corrS[8][16];

    const int tid = threadIdx.x, lane = tid & 31, w = tid >> 5;
    const int half = lane >> 4, l16 = lane & 15;
    const int h = blockIdx.y, b = blockIdx.z;
    const int bh = b * NH + h;
    const int wgbase = blockIdx.x * 128;
    const int i0 = wgbase + w * 16;

    for (int i = tid; i < SEQ; i += 256) {
        ck_pad[CKPAD + i] = ckern[(size_t)bh * SEQ + i];
        km_s[i] = km[(size_t)bh * SEQ + i];
    }
    if (tid < CKPAD) ck_pad[tid] = 0.f;
    if (tid < 128) qm_s[tid] = qm[(size_t)bh * SEQ + wgbase + tid];
    __syncthreads();

    union U16 { v16bf v; uint4 q[2]; };
    U16 ua;
    {
        const __bf16* p = qcp + ((size_t)bh * SEQ + i0 + l16) * 32;
        ua.q[0] = *(const uint4*)(p + half * 8);
        ua.q[1] = *(const uint4*)(p + 16 + half * 8);
    }
    const v16bf aq = ua.v;

    v8f c0, c1, c2, c3, zz;
    #pragma unroll
    for (int e = 0; e < 8; ++e) { c0[e] = c1[e] = c2[e] = c3[e] = zz[e] = 0.f; }
    float m_r = -1e30f, l_r = 0.f;

    // per-lane V^T row pointers for the 4 output column groups (hd = g*16 + l16)
    const __bf16* vrow0 = vT + (((size_t)bh * HD) + 0 * 16 + l16) * SEQ;
    const __bf16* vrow1 = vT + (((size_t)bh * HD) + 1 * 16 + l16) * SEQ;
    const __bf16* vrow2 = vT + (((size_t)bh * HD) + 2 * 16 + l16) * SEQ;
    const __bf16* vrow3 = vT + (((size_t)bh * HD) + 3 * 16 + l16) * SEQ;

    const int qbase = w * 16 + half * 8;            // qm_s row base for this lane
    const int nsteps = (i0 + 16 + 31) >> 5;
    for (int st = 0; st < nsteps; ++st) {
        const int j0 = st * 32;
        U16 ub0, ub1;
        {
            const __bf16* p0 = kcp + ((size_t)bh * SEQ + j0 + l16) * 32;
            ub0.q[0] = *(const uint4*)(p0 + half * 8);
            ub0.q[1] = *(const uint4*)(p0 + 16 + half * 8);
            const __bf16* p1 = kcp + ((size_t)bh * SEQ + j0 + 16 + l16) * 32;
            ub1.q[0] = *(const uint4*)(p1 + half * 8);
            ub1.q[1] = *(const uint4*)(p1 + 16 + half * 8);
        }
        v8f s0 = wmma_bf16(aq, ub0.v, zz);
        v8f s1 = wmma_bf16(aq, ub1.v, zz);

        // fold Toeplitz + qm + km + causal mask (affine DS gathers + final cndmask)
        const int ja = j0 + l16, jb = j0 + 16 + l16;
        const float kma = km_s[ja], kmb = km_s[jb & (SEQ - 1)];
        const int iB = i0 + half * 8;               // first row this lane touches
        const int baseA = CKPAD + iB - ja;          // >= CKPAD-46 > 0 within loop bounds
        const int baseB = CKPAD + iB - jb;
        #pragma unroll
        for (int r = 0; r < 8; ++r) {
            int row = r + half * 8;
            int i = iB + r;
            float base = qm_s[qbase + r];
            float va = 0.053033009f * s0[r] + ck_pad[baseA + r] + base + kma;
            float vb = 0.053033009f * s1[r] + ck_pad[baseB + r] + base + kmb;
            sS[w][row][l16]      = (ja <= i) ? va : -1e9f;
            sS[w][row][16 + l16] = (jb <= i) ? vb : -1e9f;
        }
        asm volatile("s_wait_dscnt 0" ::: "memory");

        if (half == 0) {                                    // lane l16 owns softmax row l16
            float mx = m_r;
            #pragma unroll
            for (int j = 0; j < 32; ++j) mx = fmaxf(mx, sS[w][l16][j]);
            float corr = __expf(m_r - mx);
            float sum = 0.f;
            #pragma unroll
            for (int jj = 0; jj < 16; ++jj) {               // packed pair stores
                float pa = __expf(sS[w][l16][2 * jj] - mx);
                float pb = __expf(sS[w][l16][2 * jj + 1] - mx);
                sum += pa + pb;
                union { __bf16 e2[2]; uint u; } t;
                t.e2[0] = (__bf16)pa; t.e2[1] = (__bf16)pb;
                ((uint*)&pS[w][l16][0])[jj] = t.u;
            }
            m_r = mx; l_r = l_r * corr + sum;
            corrS[w][l16] = corr;
        }
        asm volatile("s_wait_dscnt 0" ::: "memory");

        #pragma unroll
        for (int r = 0; r < 8; ++r) {                       // rescale running ctx accumulators
            float f = corrS[w][r + half * 8];
            c0[r] *= f; c1[r] *= f; c2[r] *= f; c3[r] *= f;
        }
        U16 up;
        up.q[0] = *(const uint4*)(&pS[w][l16][half * 8]);
        up.q[1] = *(const uint4*)(&pS[w][l16][16 + half * 8]);

        // V^T B-fragments: elements = contiguous keys, two b128 loads each
        const int kbase = j0 + half * 8;
        U16 uv0, uv1, uv2, uv3;
        uv0.q[0] = *(const uint4*)(vrow0 + kbase);
        uv0.q[1] = *(const uint4*)(vrow0 + kbase + 16);
        uv1.q[0] = *(const uint4*)(vrow1 + kbase);
        uv1.q[1] = *(const uint4*)(vrow1 + kbase + 16);
        uv2.q[0] = *(const uint4*)(vrow2 + kbase);
        uv2.q[1] = *(const uint4*)(vrow2 + kbase + 16);
        uv3.q[0] = *(const uint4*)(vrow3 + kbase);
        uv3.q[1] = *(const uint4*)(vrow3 + kbase + 16);
        c0 = wmma_bf16(up.v, uv0.v, c0);
        c1 = wmma_bf16(up.v, uv1.v, c1);
        c2 = wmma_bf16(up.v, uv2.v, c2);
        c3 = wmma_bf16(up.v, uv3.v, c3);
    }

    if (half == 0) corrS[w][l16] = 1.f / l_r;
    asm volatile("s_wait_dscnt 0" ::: "memory");
    #pragma unroll
    for (int r = 0; r < 8; ++r) {
        float f = corrS[w][r + half * 8];
        int i = i0 + r + half * 8;
        __bf16* op = ctxb + ((size_t)b * SEQ + i) * DMODEL + h * HD;
        op[0 * 16 + l16] = (__bf16)(c0[r] * f);
        op[1 * 16 + l16] = (__bf16)(c1[r] * f);
        op[2 * 16 + l16] = (__bf16)(c2[r] * f);
        op[3 * 16 + l16] = (__bf16)(c3[r] * f);
    }
}

// ---------------- layernorm ----------------
template <bool WB>
__global__ __launch_bounds__(256)
void ln_k(const float* __restrict__ in, const float* __restrict__ g, const float* __restrict__ bb,
          float* __restrict__ outF, __bf16* __restrict__ outB) {
    __shared__ float red[256];
    const int t = threadIdx.x;
    const float* p = in + (size_t)blockIdx.x * DMODEL;
    float s = 0.f;
    for (int i = t; i < DMODEL; i += 256) s += p[i];
    red[t] = s; __syncthreads();
    for (int o = 128; o > 0; o >>= 1) { if (t < o) red[t] += red[t + o]; __syncthreads(); }
    float mean = red[0] * (1.f / DMODEL);
    __syncthreads();
    float s2 = 0.f;
    for (int i = t; i < DMODEL; i += 256) { float d = p[i] - mean; s2 += d * d; }
    red[t] = s2; __syncthreads();
    for (int o = 128; o > 0; o >>= 1) { if (t < o) red[t] += red[t + o]; __syncthreads(); }
    float rstd = rsqrtf(red[0] * (1.f / DMODEL) + 1e-5f);
    for (int i = t; i < DMODEL; i += 256) {
        float v = (p[i] - mean) * rstd * g[i] + bb[i];
        if (outF) outF[(size_t)blockIdx.x * DMODEL + i] = v;
        if (WB)   outB[(size_t)blockIdx.x * DMODEL + i] = (__bf16)v;
    }
}

// ---------------- launch ----------------
extern "C" void kernel_launch(void* const* d_in, const int* in_sizes, int n_in,
                              void* d_out, int out_size, void* d_ws, size_t ws_size,
                              hipStream_t stream) {
    (void)in_sizes; (void)n_in; (void)out_size; (void)ws_size;
    const float* x      = (const float*)d_in[0];
    const float* v_w    = (const float*)d_in[1];
    const float* v_b    = (const float*)d_in[2];
    const float* out_w  = (const float*)d_in[3];
    const float* out_b  = (const float*)d_in[4];
    const float* qmod_w = (const float*)d_in[5];
    const float* qmod_b = (const float*)d_in[6];
    const float* kmod_w = (const float*)d_in[7];
    const float* kmod_b = (const float*)d_in[8];
    const float* gate_w = (const float*)d_in[9];
    const float* gate_b = (const float*)d_in[10];
    const float* mod_w  = (const float*)d_in[11];
    const float* mod_b  = (const float*)d_in[12];
    const float* mod_basis = (const float*)d_in[13];
    const float* freqs  = (const float*)d_in[14];
    const float* amps   = (const float*)d_in[15];
    const float* phases = (const float*)d_in[16];
    const float* cq_w   = (const float*)d_in[17];
    const float* ck_w   = (const float*)d_in[18];
    const float* ffn_w1 = (const float*)d_in[19];
    const float* ffn_b1 = (const float*)d_in[20];
    const float* ffn_w2 = (const float*)d_in[21];
    const float* ffn_b2 = (const float*)d_in[22];
    const float* ln1_g  = (const float*)d_in[23];
    const float* ln1_b  = (const float*)d_in[24];
    const float* ln2_g  = (const float*)d_in[25];
    const float* ln2_b  = (const float*)d_in[26];
    float* outp = (float*)d_out;

    char* wp = (char*)d_ws;
    auto alloc = [&](size_t bytes) -> void* {
        void* p = (void*)wp;
        wp += (bytes + 255) & ~(size_t)255;
        return p;
    };
    const size_t NTOK = (size_t)NB * SEQ;              // 2048
    __bf16* xb   = (__bf16*)alloc(NTOK * DMODEL * 2);
    __bf16* wbv  = (__bf16*)alloc((size_t)DMODEL * DMODEL * 2);
    __bf16* wbo  = (__bf16*)alloc((size_t)DMODEL * DMODEL * 2);
    __bf16* wbf1 = (__bf16*)alloc((size_t)DMODEL * 4 * DMODEL * 2);
    __bf16* wbf2 = (__bf16*)alloc((size_t)DMODEL * 4 * DMODEL * 2);
    __bf16* wbcq = (__bf16*)alloc((size_t)DMODEL * NH * NCR * 2);
    __bf16* wbck = (__bf16*)alloc((size_t)DMODEL * NH * NCR * 2);
    __bf16* vbuf = (__bf16*)alloc((size_t)NB * NH * HD * SEQ * 2);     // V^T
    __bf16* qcp  = (__bf16*)alloc((size_t)NB * NH * SEQ * 32 * 2);
    __bf16* kcp  = (__bf16*)alloc((size_t)NB * NH * SEQ * 32 * 2);
    float*  pooled = (float*)alloc((size_t)NB * DMODEL * 4);
    float*  wts    = (float*)alloc((size_t)NB * NH * NM * 4);
    float*  ckern  = (float*)alloc((size_t)NB * NH * SEQ * 4);
    float*  qmb    = (float*)alloc((size_t)NB * NH * SEQ * 4);
    float*  kmb    = (float*)alloc((size_t)NB * NH * SEQ * 4);
    __bf16* ctxb = (__bf16*)alloc(NTOK * DMODEL * 2);
    float*  hpre = (float*)alloc(NTOK * DMODEL * 4);
    float*  hf   = (float*)alloc(NTOK * DMODEL * 4);
    __bf16* hb   = (__bf16*)alloc(NTOK * DMODEL * 2);
    __bf16* ub   = (__bf16*)alloc(NTOK * 4 * DMODEL * 2);
    float*  f2   = (float*)alloc(NTOK * DMODEL * 4);

    auto cvt = [&](const float* src, __bf16* dst, int n) {
        int n4 = n / 4;
        cvt_bf16_k<<<(n4 + 255) / 256, 256, 0, stream>>>(src, dst, n4);
    };
    cvt(x, xb, (int)(NTOK * DMODEL));
    cvt(v_w, wbv, DMODEL * DMODEL);
    cvt(out_w, wbo, DMODEL * DMODEL);
    cvt(ffn_w1, wbf1, DMODEL * 4 * DMODEL);
    cvt(ffn_w2, wbf2, DMODEL * 4 * DMODEL);
    cvt(cq_w, wbcq, DMODEL * NH * NCR);
    cvt(ck_w, wbck, DMODEL * NH * NCR);
    {
        int n8 = (NB * NH * SEQ * 32) / 8;
        zero_bf16_k<<<(n8 + 255) / 256, 256, 0, stream>>>(qcp, n8);
        zero_bf16_k<<<(n8 + 255) / 256, 256, 0, stream>>>(kcp, n8);
    }

    pool_mean_k<<<(NB * DMODEL) / 256, 256, 0, stream>>>(x, pooled);
    gating_k<<<NB, 256, 0, stream>>>(pooled, gate_w, gate_b, mod_w, mod_b, mod_basis, wts);
    wave_density_k<<<(NB * NH * SEQ) / 256, 256, 0, stream>>>(freqs, amps, phases, wts, ckern);
    qkmod_k<<<NB * SEQ, 256, 0, stream>>>(x, qmod_w, qmod_b, kmod_w, kmod_b, qmb, kmb);

    // v = x@v_w + v_b   -> V^T [B,H,HD,S] bf16
    gemm_bf16_k<1><<<dim3(DMODEL / 128, (int)(NTOK / 128)), 256, 0, stream>>>(
        xb, wbv, (int)NTOK, DMODEL, DMODEL, v_b, nullptr, nullptr, vbuf);
    // qc / kc  -> padded [B,H,S,32] bf16
    gemm_bf16_k<2><<<dim3(1, (int)(NTOK / 128)), 256, 0, stream>>>(
        xb, wbcq, (int)NTOK, NH * NCR, DMODEL, nullptr, nullptr, nullptr, qcp);
    gemm_bf16_k<2><<<dim3(1, (int)(NTOK / 128)), 256, 0, stream>>>(
        xb, wbck, (int)NTOK, NH * NCR, DMODEL, nullptr, nullptr, nullptr, kcp);

    attn_k<<<dim3(SEQ / 128, NH, NB), 256, 0, stream>>>(qcp, kcp, vbuf, ckern, qmb, kmb, ctxb);

    // attn_out + residual
    gemm_bf16_k<3><<<dim3(DMODEL / 128, (int)(NTOK / 128)), 256, 0, stream>>>(
        ctxb, wbo, (int)NTOK, DMODEL, DMODEL, out_b, x, hpre, nullptr);
    ln_k<true><<<(int)NTOK, 256, 0, stream>>>(hpre, ln1_g, ln1_b, hf, hb);

    gemm_bf16_k<4><<<dim3(4 * DMODEL / 128, (int)(NTOK / 128)), 256, 0, stream>>>(
        hb, wbf1, (int)NTOK, 4 * DMODEL, DMODEL, ffn_b1, nullptr, nullptr, ub);
    gemm_bf16_k<3><<<dim3(DMODEL / 128, (int)(NTOK / 128)), 256, 0, stream>>>(
        ub, wbf2, (int)NTOK, DMODEL, 4 * DMODEL, ffn_b2, hf, f2, nullptr);
    ln_k<false><<<(int)NTOK, 256, 0, stream>>>(f2, ln2_g, ln2_b, outp, nullptr);
}
